// AttentionHead_14173392077129
// MI455X (gfx1250) — matrix operand order repaired and checked by reading, not compiled, not run on hardware
//
#include <hip/hip_runtime.h>

// ---------------------------------------------------------------------------
// Types for gfx1250 WMMA (wave32): v_wmma_f32_16x16x32_bf16
// ---------------------------------------------------------------------------
typedef __attribute__((ext_vector_type(16))) __bf16       v16bf;
typedef __attribute__((ext_vector_type(8)))  float        v8f;
typedef __attribute__((ext_vector_type(4)))  unsigned int u32x4;
typedef __attribute__((ext_vector_type(4)))  float        f32x4;

struct BF16Frag {
    union {
        v16bf        v;
        __bf16       h[16];
        unsigned int u[8];
        u32x4        q[2];
    };
};

// Native bf16 pack: lowers to v_cvt_pk_bf16_f32 on gfx1250 (RNE in hardware)
__device__ __forceinline__ unsigned int pack_bf16(float lo, float hi) {
    union { unsigned int u; __bf16 h[2]; } r;
    r.h[0] = (__bf16)lo;
    r.h[1] = (__bf16)hi;
    return r.u;
}

#define WMMA_BF16(A, B, C) \
    __builtin_amdgcn_wmma_f32_16x16x32_bf16(false, (A), false, (B), (short)0, (C), false, false)

// Problem constants
#define EMB 1024
#define HDIM 64
#define SEQ 2048
#define NB  8

// 1/sqrt(HDIM) * log2(e): folded into Q so WMMA scores are already log2-units
#define QSCALE 0.18033688011112042f

// ---------------------------------------------------------------------------
// Kernel 1: convert Wq/Wk/Wv (f32 [1024][64]) into bf16 WMMA B-fragment panels.
// Panel layout: Wp[m][kt(32)][nt(4)][lane(32)][r(8)] as u32 (bf16x2).
// B 32x16 layout: lanes 0-15 hold K=0..15 (VGPR r => K=2r,2r+1), lanes 16-31 K=16..31.
// ---------------------------------------------------------------------------
__global__ void prep_w_kernel(const float* __restrict__ Wq, const float* __restrict__ Wk,
                              const float* __restrict__ Wv, unsigned int* __restrict__ Wp) {
    const int kt = blockIdx.x;   // 0..31
    const int nt = blockIdx.y;   // 0..3
    const int m  = blockIdx.z;   // 0..2
    const float* W = (m == 0) ? Wq : (m == 1) ? Wk : Wv;
    const int l  = threadIdx.x;  // 0..31
    const int n  = nt * 16 + (l & 15);
    const int kb = kt * 32 + ((l & 16) ? 16 : 0);
    unsigned int* dst = Wp + (((size_t)m * 128 + (size_t)kt * 4 + nt) * 32 + l) * 8;
#pragma unroll
    for (int r = 0; r < 8; ++r) {
        const int k = kb + 2 * r;
        dst[r] = pack_bf16(W[(size_t)k * HDIM + n], W[(size_t)(k + 1) * HDIM + n]);
    }
}

// ---------------------------------------------------------------------------
// Kernel 2: projection GEMMs. blockIdx.y selects m: 0 -> Q=(y*Wq+bq)*QSCALE
// (row-major bf16), 1 -> K=x*Wk+bk (row-major bf16), 2 -> V=x*Wv+bv stored
// TRANSPOSED Vt[b][h][s] bf16. One wave: 16-row x 64-col tile, K=1024 / 32 steps.
// ---------------------------------------------------------------------------
__global__ void proj_kernel(const float* __restrict__ x, const float* __restrict__ y,
                            const float* __restrict__ bq, const float* __restrict__ bk,
                            const float* __restrict__ bv, const unsigned int* __restrict__ Wp,
                            unsigned short* __restrict__ Qbf, unsigned short* __restrict__ Kbf,
                            unsigned short* __restrict__ Vt) {
    const int m  = blockIdx.y;
    const int r0 = blockIdx.x * 16;
    const float* src  = (m == 0) ? y : x;
    const float* bias = (m == 0) ? bq : (m == 1) ? bk : bv;
    const unsigned int* Wm = Wp + (size_t)m * 128 * 32 * 8;
    const int  l  = threadIdx.x;
    const bool lo = (l < 16);
    const int  row = r0 + (l & 15);

    v8f c[4];
#pragma unroll
    for (int t = 0; t < 4; ++t) {
        const float bb = bias[16 * t + (l & 15)];
#pragma unroll
        for (int i = 0; i < 8; ++i) c[t][i] = bb;
    }

    const float* ap = src + (size_t)row * EMB;
    for (int ks = 0; ks < 32; ++ks) {
        // A-fragment 16x32 bf16: lane holds row (l&15); K chunk base..base+15 contiguous
        const int kb1 = ks * 32 + (lo ? 0 : 16);
        f32x4 a0 = *(const f32x4*)(ap + kb1);
        f32x4 a1 = *(const f32x4*)(ap + kb1 + 4);
        f32x4 a2 = *(const f32x4*)(ap + kb1 + 8);
        f32x4 a3 = *(const f32x4*)(ap + kb1 + 12);
        BF16Frag a;
#pragma unroll
        for (int j = 0; j < 4; ++j) {
            a.h[j]      = (__bf16)a0[j];
            a.h[4 + j]  = (__bf16)a1[j];
            a.h[8 + j]  = (__bf16)a2[j];
            a.h[12 + j] = (__bf16)a3[j];
        }
#pragma unroll
        for (int nt = 0; nt < 4; ++nt) {
            BF16Frag bw;
            const u32x4* wp4 = (const u32x4*)(Wm + (((size_t)ks * 4 + nt) * 32 + l) * 8);
            bw.q[0] = wp4[0]; bw.q[1] = wp4[1];
            c[nt] = WMMA_BF16(a.v, bw.v, c[nt]);
        }
    }

    // C layout: lane n=(l&15) -> col h=16t+(l&15); vgpr i -> row r0+i+(lo?0:8)
    if (m == 0) {
        __bf16* D = (__bf16*)Qbf;
#pragma unroll
        for (int t = 0; t < 4; ++t) {
            const int h = 16 * t + (l & 15);
#pragma unroll
            for (int i = 0; i < 8; ++i) {
                const int orow = r0 + i + (lo ? 0 : 8);
                D[(size_t)orow * HDIM + h] = (__bf16)(c[t][i] * QSCALE);
            }
        }
    } else if (m == 1) {
        __bf16* D = (__bf16*)Kbf;
#pragma unroll
        for (int t = 0; t < 4; ++t) {
            const int h = 16 * t + (l & 15);
#pragma unroll
            for (int i = 0; i < 8; ++i) {
                const int orow = r0 + i + (lo ? 0 : 8);
                D[(size_t)orow * HDIM + h] = (__bf16)c[t][i];
            }
        }
    } else {
        const int b  = r0 >> 11;        // r0 / 2048
        const int sr = r0 & 2047;
#pragma unroll
        for (int t = 0; t < 4; ++t) {
            const int h = 16 * t + (l & 15);
            union { u32x4 q; __bf16 hh[8]; } pk;
#pragma unroll
            for (int i = 0; i < 8; ++i) pk.hh[i] = (__bf16)c[t][i];
            *(u32x4*)(Vt + ((size_t)(b * HDIM + h) * SEQ + sr + (lo ? 0 : 8))) = pk.q;
        }
    }
}

// ---------------------------------------------------------------------------
// Kernel 3: suffix sums of V: SV[b][q][h] = sum_{k>q} V[b][k][h] (fp32),
// read from transposed Vt rows (contiguous). One block per (b,h).
// ---------------------------------------------------------------------------
__global__ void suffix_kernel(const unsigned short* __restrict__ Vt, float* __restrict__ SV) {
    const int b = blockIdx.x >> 6;
    const int h = blockIdx.x & 63;
    const unsigned short* row = Vt + (size_t)(b * HDIM + h) * SEQ;
    const int t = threadIdx.x;   // 256 threads, 8 elements each
    union { u32x4 q; __bf16 hh[8]; } raw;
    raw.q = *(const u32x4*)(row + t * 8);
    float v[8];
#pragma unroll
    for (int j = 0; j < 8; ++j) v[j] = (float)raw.hh[j];
    float cs = 0.f;
#pragma unroll
    for (int j = 0; j < 8; ++j) cs += v[j];
    __shared__ float s[256];
    s[t] = cs;
    __syncthreads();
    for (int off = 1; off < 256; off <<= 1) {
        float add = (t >= off) ? s[t - off] : 0.f;
        __syncthreads();
        s[t] += add;
        __syncthreads();
    }
    const float incl  = s[t];
    const float total = s[255];
    float run = incl - cs;   // exclusive prefix for this chunk
#pragma unroll
    for (int j = 0; j < 8; ++j) {
        run += v[j];
        SV[(size_t)(b * SEQ + t * 8 + j) * HDIM + h] = total - run;
    }
}

// ---------------------------------------------------------------------------
// Kernel 4: flash attention, exact pre-softmax tril(0) semantics, base-2 softmax.
// scoresT = K*Q^T (q on lanes), O^T = V^T*P^T (q on lanes -> per-lane stats).
// Masked region (k>q contributes 2^0 each) handled via count + suffix sums SV.
// ---------------------------------------------------------------------------
__global__ void __launch_bounds__(128) flash_kernel(
        const unsigned short* __restrict__ Qbf, const unsigned short* __restrict__ Kbf,
        const unsigned short* __restrict__ Vt, const float* __restrict__ SV,
        float* __restrict__ out) {
    const int  b  = blockIdx.y;
    const int  w  = threadIdx.x >> 5;
    const int  l  = threadIdx.x & 31;
    const bool lo = (l < 16);
    const int  qt = blockIdx.x * 4 + w;
    const int  qb = qt * 16;
    const int  q  = qb + (l & 15);
    const unsigned short* Qb = Qbf + (size_t)b * SEQ * HDIM;
    const unsigned short* Kb = Kbf + (size_t)b * SEQ * HDIM;
    const unsigned short* Vb = Vt  + (size_t)b * HDIM * SEQ;

    // B-fragments of Q^T (lane holds Q row q, contiguous h): loaded once
    BF16Frag bq2[2];
#pragma unroll
    for (int hh = 0; hh < 2; ++hh) {
        const int hb = 32 * hh + (lo ? 0 : 16);
        const u32x4* p = (const u32x4*)(Qb + (size_t)q * HDIM + hb);
        bq2[hh].q[0] = p[0]; bq2[hh].q[1] = p[1];
    }

    v8f acc[4];
#pragma unroll
    for (int t = 0; t < 4; ++t)
#pragma unroll
        for (int i = 0; i < 8; ++i) acc[t][i] = 0.f;
    float m = 0.f, lr = 0.f;        // log2-domain max starts at 0 (masked zeros)
    const int nch = qt / 2 + 1;     // 32-key chunks covering k <= qb+15

    for (int ch = 0; ch < nch; ++ch) {
        const int cb = ch * 32;
        v8f c0, c1;
#pragma unroll
        for (int i = 0; i < 8; ++i) { c0[i] = 0.f; c1[i] = 0.f; }
        // scoresT tiles: A = K rows (16 keys x 32 h), B = Q^T (pre-scaled by log2e/8)
#pragma unroll
        for (int s2 = 0; s2 < 2; ++s2) {
            const int key = cb + 16 * s2 + (l & 15);
            const unsigned short* kp = Kb + (size_t)key * HDIM;
#pragma unroll
            for (int hh = 0; hh < 2; ++hh) {
                const int hb1 = 32 * hh + (lo ? 0 : 16);
                BF16Frag a;
                a.q[0] = *(const u32x4*)(kp + hb1);
                a.q[1] = *(const u32x4*)(kp + hb1 + 8);
                if (s2 == 0) c0 = WMMA_BF16(a.v, bq2[hh].v, c0);
                else         c1 = WMMA_BF16(a.v, bq2[hh].v, c1);
            }
        }
        // lane l holds q=(qb+l%16); c0 vgpr i -> key cb+i(+8 hi-half); c1 -> +16
        float sc0[8], sc1[8];
        float tmax = -3.0e38f;
#pragma unroll
        for (int i = 0; i < 8; ++i) {
            const int k0 = cb + i + (lo ? 0 : 8);
            const int k1 = k0 + 16;
            sc0[i] = (k0 <= q) ? c0[i] : -1.0e30f;
            sc1[i] = (k1 <= q) ? c1[i] : -1.0e30f;
            tmax = fmaxf(tmax, fmaxf(sc0[i], sc1[i]));
        }
        tmax = fmaxf(tmax, __shfl_xor(tmax, 16));
        const float nm = fmaxf(m, tmax);
        const float r  = exp2f(m - nm);
        float p0[8], p1[8], ts = 0.f;
#pragma unroll
        for (int i = 0; i < 8; ++i) {
            p0[i] = exp2f(sc0[i] - nm);
            p1[i] = exp2f(sc1[i] - nm);
            ts += p0[i] + p1[i];
        }
        ts += __shfl_xor(ts, 16);
        lr = lr * r + ts;
        m = nm;
#pragma unroll
        for (int t = 0; t < 4; ++t)
#pragma unroll
            for (int i = 0; i < 8; ++i) acc[t][i] *= r;   // per-lane scalar rescale

        // Build P^T B-fragment (lane: col q, K=keys) via one cross-half exchange
        unsigned int pk0[4], pk1[4], o0[4], o1[4];
#pragma unroll
        for (int j = 0; j < 4; ++j) {
            pk0[j] = pack_bf16(p0[2 * j], p0[2 * j + 1]);
            pk1[j] = pack_bf16(p1[2 * j], p1[2 * j + 1]);
        }
#pragma unroll
        for (int j = 0; j < 4; ++j) {
            o0[j] = (unsigned int)__shfl_xor((int)pk0[j], 16);
            o1[j] = (unsigned int)__shfl_xor((int)pk1[j], 16);
        }
        BF16Frag bp;
#pragma unroll
        for (int j = 0; j < 4; ++j) {
            bp.u[j]     = lo ? pk0[j] : o1[j];   // K 0..7  / 16..23
            bp.u[4 + j] = lo ? o0[j]  : pk1[j];  // K 8..15 / 24..31
        }
        // O^T += V^T * P^T ; A = V^T rows (16 h x 32 keys) from transposed Vt
#pragma unroll
        for (int t = 0; t < 4; ++t) {
            const int h = 16 * t + (l & 15);
            const unsigned short* vp = Vb + (size_t)h * SEQ + cb + (lo ? 0 : 16);
            BF16Frag a;
            a.q[0] = *(const u32x4*)vp;
            a.q[1] = *(const u32x4*)(vp + 8);
            acc[t] = WMMA_BF16(a.v, bp.v, acc[t]);
        }
    }

    // Exact masked-region contribution: (S-1-q) zero-scores + 2^{-m} * SV[q]
    const float e0 = exp2f(-m);
    lr += (float)(SEQ - 1 - q) * e0;
    const float inv = 1.0f / lr;
    const float* svp = SV  + (size_t)(b * SEQ + q) * HDIM;
    float*       op  = out + (size_t)(b * SEQ + q) * HDIM;
#pragma unroll
    for (int t = 0; t < 4; ++t) {
        const int hb = 16 * t + (lo ? 0 : 8);   // vgpr i -> h = hb + i (contiguous)
        f32x4 s0 = *(const f32x4*)(svp + hb);
        f32x4 s1 = *(const f32x4*)(svp + hb + 4);
        f32x4 r0, r1;
#pragma unroll
        for (int i = 0; i < 4; ++i) {
            r0[i] = (acc[t][i]     + e0 * s0[i]) * inv;
            r1[i] = (acc[t][4 + i] + e0 * s1[i]) * inv;
        }
        *(f32x4*)(op + hb)     = r0;
        *(f32x4*)(op + hb + 4) = r1;
    }
}

// ---------------------------------------------------------------------------
extern "C" void kernel_launch(void* const* d_in, const int* in_sizes, int n_in,
                              void* d_out, int out_size, void* d_ws, size_t ws_size,
                              hipStream_t stream) {
    const float* x  = (const float*)d_in[0];
    const float* y  = (const float*)d_in[1];
    const float* Wq = (const float*)d_in[2];
    const float* bq = (const float*)d_in[3];
    const float* Wk = (const float*)d_in[4];
    const float* bk = (const float*)d_in[5];
    const float* Wv = (const float*)d_in[6];
    const float* bv = (const float*)d_in[7];
    float* out = (float*)d_out;

    char* ws = (char*)d_ws;
    unsigned short* Qbf = (unsigned short*)(ws);                         // 2 MB
    unsigned short* Kbf = (unsigned short*)(ws + (2ull << 20));          // 2 MB
    unsigned short* Vt  = (unsigned short*)(ws + (4ull << 20));          // 2 MB
    float*          SV  = (float*)(ws + (6ull << 20));                   // 4 MB
    unsigned int*   Wp  = (unsigned int*)(ws + (10ull << 20));           // 384 KB

    prep_w_kernel<<<dim3(32, 4, 3), 32, 0, stream>>>(Wq, Wk, Wv, Wp);
    proj_kernel<<<dim3((NB * SEQ) / 16, 3), 32, 0, stream>>>(x, y, bq, bk, bv, Wp, Qbf, Kbf, Vt);
    suffix_kernel<<<NB * HDIM, 256, 0, stream>>>(Vt, SV);
    flash_kernel<<<dim3(SEQ / 64, NB), 128, 0, stream>>>(Qbf, Kbf, Vt, SV, out);
}
